// Feature_extr_31997506355472
// MI455X (gfx1250) — matrix operand order repaired and checked by reading, not compiled, not run on hardware
//
#include <hip/hip_runtime.h>
#include <hip/hip_bf16.h>

typedef __attribute__((ext_vector_type(16))) _Float16 v16h;
typedef __attribute__((ext_vector_type(8)))  _Float16 v8h;
typedef __attribute__((ext_vector_type(8)))  float    v8f;

#if defined(__has_builtin)
# if __has_builtin(__builtin_amdgcn_tensor_load_to_lds) && __has_builtin(__builtin_amdgcn_s_wait_tensorcnt)
#  define USE_TDM 1
# endif
#endif
#ifndef USE_TDM
# define USE_TDM 0
#endif

#if USE_TDM
typedef __attribute__((ext_vector_type(4))) unsigned int tdm_v4u;
typedef __attribute__((ext_vector_type(8))) int          tdm_v8i;
typedef __attribute__((ext_vector_type(4))) int          tdm_v4i;
#endif

#define NT 4   // N-tiles (of 16 channels) per block => 64 output channels

// ---------------------------------------------------------------------------
// 3D convolution as LDS-staged implicit GEMM on v_wmma_f32_16x16x32_f16.
// x: [Cin, ID, IH, IW] (f16)   w: [Cout, Cin*KD*KH*KW] (f16)
// y: [Cout, OD, OH, OW] (f32)
// Block: 256 threads = 8 waves; block tile = 128 (M) x 64 (N).
// Per K-step(32): TDM DMA of the 32x64 weight tile into LDS (issued first,
// overlapping the gather), cooperative branchless im2col gather of A into
// LDS, then per-wave ds_load_b128 fragments and 4 WMMAs.
// ---------------------------------------------------------------------------
__global__ __launch_bounds__(256) void conv3d_wmma(
    const _Float16* __restrict__ x, const _Float16* __restrict__ w,
    float* __restrict__ y,
    int Cin, int Cout,
    int ID, int IH, int IW,
    int OD, int OH, int OW,
    int KD, int KH, int KW,
    int sd, int sh, int sw,
    int pd, int ph, int pw)
{
    __shared__ int kmap[5888];                       // max K = 8*7*7*15 = 5880
    __shared__ int mcoord[128];                      // packed (bd+8,bh+8,bw+8)
    __shared__ int mbase[128];                       // voxel base offset (elems)
    __shared__ __attribute__((aligned(16))) _Float16 As[128 * 32];     // 8 KB
    __shared__ __attribute__((aligned(16))) _Float16 Bs[NT * 16 * 32]; // 4 KB

    const int tid  = threadIdx.x;
    const int KHW  = KH * KW;
    const int KDHW = KD * KHW;
    const int K    = Cin * KDHW;

    for (int k = tid; k < K; k += 256) {
        int ci = k / KDHW;
        int r  = k - ci * KDHW;
        int kd = r / KHW;   r -= kd * KHW;
        int kh = r / KW;
        int kw = r - kh * KW;
        kmap[k] = (ci << 15) | (kd << 10) | (kh << 5) | kw;
    }

    const long M  = (long)OD * OH * OW;
    const long mB = (long)blockIdx.x * 128;
    if (tid < 128) {
        long m = mB + tid; if (m >= M) m = M - 1;
        int od = (int)(m / (OH * OW));
        int r2 = (int)(m - (long)od * (OH * OW));
        int oh = r2 / OW;
        int ow = r2 - oh * OW;
        int bd = od * sd - pd;
        int bh = oh * sh - ph;
        int bw = ow * sw - pw;
        mcoord[tid] = ((bd + 8) << 20) | ((bh + 8) << 10) | (bw + 8);
        mbase[tid]  = (bd * IH + bh) * IW + bw;      // may be negative (guarded)
    }

    const int lane = tid & 31;
    const int wave = tid >> 5;
    const int hi   = lane >> 4;
    const int ml   = lane & 15;
    const int n0   = blockIdx.y * (NT * 16);

    const int tK = tid & 31;      // staging: k within tile
    const int tM = tid >> 5;      // staging: base m row (8 rows/pass)

    v8f acc[NT] = {};
    for (int k0 = 0; k0 < K; k0 += 32) {
        __syncthreads();          // previous fragments consumed / tables ready

        // ---------- B tile DMA first (overlaps the A gather) --------------
#if USE_TDM
        if (tid < 32) {                          // wave 0 issues the DMA
            unsigned long long ga =
                (unsigned long long)(uintptr_t)(w + (size_t)n0 * K + k0);
            unsigned int ldsOff =
                (unsigned int)((uintptr_t)&Bs[0] & 0xFFFFFFFFull);
            tdm_v4u g0;
            g0[0] = 1u;                                          // count=1
            g0[1] = ldsOff;                                      // lds_addr
            g0[2] = (unsigned int)ga;                            // global[31:0]
            g0[3] = ((unsigned int)(ga >> 32) & 0x01FFFFFFu)
                  | 0x80000000u;                                 // type=2
            tdm_v8i g1;
            g1[0] = (int)(1u << 16);                             // data_size=2B
            g1[1] = (int)(((unsigned)K & 0xFFFFu) << 16);        // tdim0 lo
            g1[2] = (int)((((unsigned)K >> 16) & 0xFFFFu)
                  | (((unsigned)Cout & 0xFFFFu) << 16));         // tdim0 hi|tdim1 lo
            g1[3] = (int)((32u << 16));                          // tile_dim0=32
            g1[4] = (int)(unsigned)(NT * 16);                    // tile_dim1=64
            g1[5] = (int)(unsigned)K;                            // stride0 (elems)
            g1[6] = 0;
            g1[7] = 0;
            tdm_v4i g2 = {0, 0, 0, 0};
            tdm_v4i g3 = {0, 0, 0, 0};
            tdm_v8i g4 = {0, 0, 0, 0, 0, 0, 0, 0};
            __builtin_amdgcn_tensor_load_to_lds(g0, g1, g2, g3, g4, 0);
        }
#else
        for (int i = tid; i < NT * 16 * 32; i += 256) {
            int n = i >> 5, k = i & 31;
            int gk = k0 + k, gn = n0 + n;
            _Float16 v = (gk < K && gn < Cout) ? w[(size_t)gn * K + gk]
                                               : (_Float16)0.0f;
            Bs[i] = v;
        }
#endif

        // ---------- stage A tile (128 x 32) : branchless im2col gather ----
        const int  kk  = k0 + tK;
        const bool kok = (kk < K);
        int pk = kmap[kok ? kk : 0];
        int kw_ =  pk        & 31;
        int kh_ = (pk >> 5)  & 31;
        int kd_ = (pk >> 10) & 31;
        int ci  =  pk >> 15;
        int delta = ((ci * ID + kd_) * IH + kh_) * IW + kw_;
#pragma unroll
        for (int p = 0; p < 16; ++p) {
            int mi = p * 8 + tM;
            int pc = mcoord[mi];
            int bb = mbase[mi];
            int id = ((pc >> 20) & 1023) - 8 + kd_;
            int ih = ((pc >> 10) & 1023) - 8 + kh_;
            int iw = ( pc        & 1023) - 8 + kw_;
            bool inb = kok & ((unsigned)id < (unsigned)ID)
                           & ((unsigned)ih < (unsigned)IH)
                           & ((unsigned)iw < (unsigned)IW);
            int off  = bb + delta;
            int offs = inb ? off : 0;           // clamp: unconditional load
            _Float16 v = x[offs];
            v = inb ? v : (_Float16)0.0f;       // v_cndmask, no branch
            As[mi * 32 + tK] = v;
        }

#if USE_TDM
        if (tid < 32) __builtin_amdgcn_s_wait_tensorcnt(0);
#endif
        __syncthreads();

        // ---------- fragments + 4 WMMAs -----------------------------------
        const v8h* ap = (const v8h*)&As[(wave * 16 + ml) * 32];
        v8h a0 = ap[hi];            // halves [hi*8 .. hi*8+7]
        v8h a1 = ap[2 + hi];        // halves [16+hi*8 ..]
        v16h a;
#pragma unroll
        for (int e = 0; e < 8; ++e) { a[e] = a0[e]; a[e + 8] = a1[e]; }
#pragma unroll
        for (int nt = 0; nt < NT; ++nt) {
            const v8h* bp = (const v8h*)&Bs[(nt * 16 + ml) * 32];
            v8h b0 = bp[hi];            // k = hi*8 + 0..7
            v8h b1 = bp[2 + hi];        // k = 16 + hi*8 + 0..7
            v16h b;
#pragma unroll
            for (int e = 0; e < 8; ++e) { b[e] = b0[e]; b[e + 8] = b1[e]; }
            acc[nt] = __builtin_amdgcn_wmma_f32_16x16x32_f16(
                false, a, false, b, (short)0, acc[nt], false, false);
        }
    }

    // D layout: VGPR r -> M = r + 8*hi, N = ml
    const long m0 = mB + wave * 16;
#pragma unroll
    for (int nt = 0; nt < NT; ++nt) {
        int cn = n0 + nt * 16 + ml;
        if (cn < Cout) {
            float* yrow = y + (size_t)cn * M;
#pragma unroll
            for (int rr = 0; rr < 8; ++rr) {
                long mm = m0 + rr + (hi ? 8 : 0);
                if (mm < M) yrow[mm] = acc[nt][rr];
            }
        }
    }
}

// ---------------------------------------------------------------------------
// Support kernels
// ---------------------------------------------------------------------------
__global__ void cvt_f16(const float* __restrict__ s, _Float16* __restrict__ d, long n) {
    long i = (long)blockIdx.x * blockDim.x + threadIdx.x;
    if (i < n) d[i] = (_Float16)s[i];
}

__global__ void zero_f32(float* p, long n) {
    long i = (long)blockIdx.x * blockDim.x + threadIdx.x;
    if (i < n) p[i] = 0.0f;
}

__global__ void scatter_kernel(const int* __restrict__ coords,
                               const float* __restrict__ feats,
                               float* grid, float* cnt, int n, int DY, int DZ) {
    int i = blockIdx.x * blockDim.x + threadIdx.x;
    if (i >= n) return;
    int xx = coords[i * 4 + 0], yy = coords[i * 4 + 1], zz = coords[i * 4 + 2];
    long idx = ((long)xx * DY + yy) * DZ + zz;
    atomicAdd(grid + idx, feats[i]);
    atomicAdd(cnt + idx, 1.0f);
}

__global__ void mask_from_cnt(float* m, long n) {
    long i = (long)blockIdx.x * blockDim.x + threadIdx.x;
    if (i < n) m[i] = (m[i] > 0.0f) ? 1.0f : 0.0f;
}

__global__ void mask_mul(float* y, const float* __restrict__ mask, int C, long V) {
    long v = (long)blockIdx.x * blockDim.x + threadIdx.x;
    if (v >= V) return;
    float m = mask[v];
    for (int c = 0; c < C; ++c) y[(long)c * V + v] *= m;
}

__global__ void add_k(float* y, const float* __restrict__ r, long n) {
    long i = (long)blockIdx.x * blockDim.x + threadIdx.x;
    if (i < n) y[i] += r[i];
}

__global__ void reduce_sum1(const float* __restrict__ p, long n, float* out) {
    __shared__ float s[1024];
    float acc = 0.0f;
    for (long i = threadIdx.x; i < n; i += blockDim.x) acc += p[i];
    s[threadIdx.x] = acc; __syncthreads();
    for (int st = 512; st > 0; st >>= 1) {
        if ((int)threadIdx.x < st) s[threadIdx.x] += s[threadIdx.x + st];
        __syncthreads();
    }
    if (threadIdx.x == 0) out[0] = s[0];
}

__global__ void bn_stats(const float* __restrict__ x, long V, float* stats) {
    int c = blockIdx.x;
    const float* p = x + (size_t)c * V;
    __shared__ float s0[256], s1[256];
    float a = 0.0f, b = 0.0f;
    for (long i = threadIdx.x; i < V; i += 256) { float v = p[i]; a += v; b += v * v; }
    s0[threadIdx.x] = a; s1[threadIdx.x] = b; __syncthreads();
    for (int st = 128; st > 0; st >>= 1) {
        if ((int)threadIdx.x < st) {
            s0[threadIdx.x] += s0[threadIdx.x + st];
            s1[threadIdx.x] += s1[threadIdx.x + st];
        }
        __syncthreads();
    }
    if (threadIdx.x == 0) { stats[2 * c] = s0[0]; stats[2 * c + 1] = s1[0]; }
}

__global__ void bn_apply(float* x, const float* __restrict__ mask,
                         const float* __restrict__ g, const float* __restrict__ b,
                         const float* __restrict__ stats, const float* __restrict__ cntp,
                         long V) {
    int  c = blockIdx.y;
    long v = (long)blockIdx.x * blockDim.x + threadIdx.x;
    if (v >= V) return;
    float cnt  = fmaxf(cntp[0], 1.0f);
    float mean = stats[2 * c] / cnt;
    float var  = stats[2 * c + 1] / cnt - mean * mean;
    float inv  = g[c] * rsqrtf(var + 1e-5f);
    long  i    = (long)c * V + v;
    float yv   = x[i] * inv + (b[c] - mean * inv);
    x[i] = fmaxf(yv, 0.0f) * mask[v];
}

__global__ void max_mask_k(const float* __restrict__ mi, float* mo,
                           int ID, int IH, int IW, int OD, int OH, int OW,
                           int KD, int KH, int KW, int sd, int sh, int sw) {
    long o = (long)blockIdx.x * blockDim.x + threadIdx.x;
    long M = (long)OD * OH * OW;
    if (o >= M) return;
    int od = (int)(o / (OH * OW));
    int r  = (int)(o - (long)od * (OH * OW));
    int oh = r / OW, ow = r - (r / OW) * OW;
    int bd = od * sd, bh = oh * sh, bw = ow * sw;
    float v = 0.0f;
    for (int kd = 0; kd < KD; ++kd)
        for (int kh = 0; kh < KH; ++kh)
            for (int kw = 0; kw < KW; ++kw) {
                float m = mi[((size_t)(bd + kd) * IH + (bh + kh)) * IW + (bw + kw)];
                v = fmaxf(v, m);
            }
    mo[o] = (v > 0.0f) ? 1.0f : 0.0f;
}

__global__ void final_pool(const float* __restrict__ x, float* out) {
    int c = threadIdx.x;   // 256 threads
    float s = 0.0f;
    for (int j = 0; j < 8; ++j) s += x[c * 8 + j];
    out[c] = s * 0.125f;
}

// ---------------------------------------------------------------------------
// Host orchestration
// ---------------------------------------------------------------------------
extern "C" void kernel_launch(void* const* d_in, const int* in_sizes, int n_in,
                              void* d_out, int out_size, void* d_ws, size_t ws_size,
                              hipStream_t stream) {
    (void)in_sizes; (void)n_in; (void)out_size; (void)ws_size;
    const int DXc = 99, DYc = 99, DZc = 199;
    const long V0 = (long)DXc * DYc * DZc;          // 1,950,399
    const long SBUF = 1700000;                      // rotating small buffers

    // ---- input pointers (setup_inputs insertion order) ----
    int ii = 0;
    const int*   coords = (const int*)d_in[ii++];
    const float* feats  = (const float*)d_in[ii++];
    const float* w_in1  = (const float*)d_in[ii++];
    const float* w_in2  = (const float*)d_in[ii++];
    const float* w_down = (const float*)d_in[ii++];
    const float* bn1g = (const float*)d_in[ii++]; const float* bn1b = (const float*)d_in[ii++];
    const float* bn2g = (const float*)d_in[ii++]; const float* bn2b = (const float*)d_in[ii++];
    struct Blk { const float *bn0g,*bn0b,*w1,*bn1g,*bn1b,*w2,*wres; int nin,n,s; };
    const int BC[8][3] = {{16,16,1},{16,32,2},{32,32,1},{32,64,2},
                          {64,64,1},{64,128,2},{128,128,1},{128,256,2}};
    Blk blk[8];
    for (int bI = 0; bI < 8; ++bI) {
        blk[bI].nin = BC[bI][0]; blk[bI].n = BC[bI][1]; blk[bI].s = BC[bI][2];
        blk[bI].bn0g = (const float*)d_in[ii++]; blk[bI].bn0b = (const float*)d_in[ii++];
        blk[bI].w1   = (const float*)d_in[ii++];
        blk[bI].bn1g = (const float*)d_in[ii++]; blk[bI].bn1b = (const float*)d_in[ii++];
        blk[bI].w2   = (const float*)d_in[ii++];
        blk[bI].wres = (blk[bI].s > 1) ? (const float*)d_in[ii++] : nullptr;
    }
    const float* bnfg = (const float*)d_in[ii++];
    const float* bnfb = (const float*)d_in[ii++];

    // ---- workspace layout ----
    float* W     = (float*)d_ws;
    float* grid  = W;                  // V0
    float* mask0 = grid  + V0;         // V0 (counts -> mask in place)
    float* A8    = mask0 + V0;         // 8*V0
    float* B8    = A8 + 8 * V0;        // 8*V0
    float* S[4];
    S[0] = B8 + 8 * V0;
    S[1] = S[0] + SBUF; S[2] = S[1] + SBUF; S[3] = S[2] + SBUF;
    float* m1 = S[3] + SBUF;           // 47^3
    float* m2 = m1 + 103823;           // 23^3
    float* m3 = m2 + 12167;            // 11^3
    float* m4 = m3 + 1331;             // 5^3
    float* m5 = m4 + 125;              // 2^3
    float* stats  = m5 + 8;            // 512
    float* cntbuf = stats + 512;       // 1
    _Float16* xf16 = (_Float16*)(cntbuf + 1);          // 16M halves (8M floats)
    _Float16* wf16 = (_Float16*)((float*)xf16 + 8000000);  // 128K halves

    auto zero = [&](float* p, long n) {
        zero_f32<<<dim3((unsigned)((n + 255) / 256)), 256, 0, stream>>>(p, n);
    };
    auto conv = [&](const float* x, const float* w, float* y, int Cin, int Cout,
                    int ID, int IH, int IW, int OD, int OH, int OW,
                    int KD, int KH, int KW, int sd, int sh, int sw,
                    int pd, int ph, int pw) {
        long nx = (long)Cin * ID * IH * IW;
        long nw = (long)Cout * Cin * KD * KH * KW;
        cvt_f16<<<dim3((unsigned)((nx + 255) / 256)), 256, 0, stream>>>(x, xf16, nx);
        cvt_f16<<<dim3((unsigned)((nw + 255) / 256)), 256, 0, stream>>>(w, wf16, nw);
        long M = (long)OD * OH * OW;
        dim3 g((unsigned)((M + 127) / 128), (unsigned)((Cout + NT * 16 - 1) / (NT * 16)));
        conv3d_wmma<<<g, 256, 0, stream>>>(xf16, wf16, y, Cin, Cout, ID, IH, IW,
                                           OD, OH, OW, KD, KH, KW, sd, sh, sw, pd, ph, pw);
    };
    auto mmul = [&](float* y, const float* mk, int C, long V) {
        mask_mul<<<dim3((unsigned)((V + 255) / 256)), 256, 0, stream>>>(y, mk, C, V);
    };
    auto bnrelu = [&](float* x, const float* mk, const float* g, const float* b,
                      int C, long V) {
        reduce_sum1<<<1, 1024, 0, stream>>>(mk, V, cntbuf);
        bn_stats<<<dim3((unsigned)C), 256, 0, stream>>>(x, V, stats);
        dim3 gg((unsigned)((V + 255) / 256), (unsigned)C);
        bn_apply<<<gg, 256, 0, stream>>>(x, mk, g, b, stats, cntbuf, V);
    };
    auto addv = [&](float* y, const float* r, long n) {
        add_k<<<dim3((unsigned)((n + 255) / 256)), 256, 0, stream>>>(y, r, n);
    };
    auto maxmask = [&](const float* mi, float* mo, int ID, int IH, int IW,
                       int OD, int OH, int OW, int KD, int KH, int KW,
                       int sd, int sh, int sw) {
        long M = (long)OD * OH * OW;
        max_mask_k<<<dim3((unsigned)((M + 255) / 256)), 256, 0, stream>>>(
            mi, mo, ID, IH, IW, OD, OH, OW, KD, KH, KW, sd, sh, sw);
    };

    // ---- 1. scatter points into grid, build mask ----
    zero(grid, V0); zero(mask0, V0);
    scatter_kernel<<<dim3((10000 + 255) / 256), 256, 0, stream>>>(
        coords, feats, grid, mask0, 10000, DYc, DZc);
    mask_from_cnt<<<dim3((unsigned)((V0 + 255) / 256)), 256, 0, stream>>>(mask0, V0);

    // ---- 2. stem: two 7x7x15 convs at full res ----
    conv(grid, w_in1, A8, 1, 8, DXc, DYc, DZc, DXc, DYc, DZc, 7, 7, 15, 1, 1, 1, 3, 3, 7);
    mmul(A8, mask0, 8, V0);
    bnrelu(A8, mask0, bn1g, bn1b, 8, V0);
    conv(A8, w_in2, B8, 8, 8, DXc, DYc, DZc, DXc, DYc, DZc, 7, 7, 15, 1, 1, 1, 3, 3, 7);
    mmul(B8, mask0, 8, V0);
    bnrelu(B8, mask0, bn2g, bn2b, 8, V0);

    // ---- 3. downsample to 47^3 x 16 ----
    conv(B8, w_down, S[0], 8, 16, DXc, DYc, DZc, 47, 47, 47, 7, 7, 15, 2, 2, 4, 0, 0, 0);
    maxmask(mask0, m1, DXc, DYc, DZc, 47, 47, 47, 7, 7, 15, 2, 2, 4);

    // ---- 4. residual blocks ----
    float* mcur = m1;
    float* mnext[4] = { m2, m3, m4, m5 };
    int mni = 0;
    int xi = 0;
    int dd = 47;
    for (int bI = 0; bI < 8; ++bI) {
        Blk& p = blk[bI];
        long Vin = (long)dd * dd * dd;
        int o1 = (xi + 1) & 3, o2 = (xi + 2) & 3, o3 = (xi + 3) & 3;
        if (p.s == 1) {
            bnrelu(S[xi], mcur, p.bn0g, p.bn0b, p.nin, Vin);            // xb (= residual)
            conv(S[xi], p.w1, S[o1], p.nin, p.n, dd, dd, dd, dd, dd, dd,
                 3, 3, 3, 1, 1, 1, 1, 1, 1);
            mmul(S[o1], mcur, p.n, Vin);
            bnrelu(S[o1], mcur, p.bn1g, p.bn1b, p.n, Vin);
            conv(S[o1], p.w2, S[o2], p.n, p.n, dd, dd, dd, dd, dd, dd,
                 3, 3, 3, 1, 1, 1, 1, 1, 1);
            mmul(S[o2], mcur, p.n, Vin);
            addv(S[o2], S[xi], (long)p.n * Vin);
            xi = o2;
        } else {
            int nd = (dd - 3) / 2 + 1;
            long Vout = (long)nd * nd * nd;
            bnrelu(S[xi], mcur, p.bn0g, p.bn0b, p.nin, Vin);            // xb
            conv(S[xi], p.w1, S[o1], p.nin, p.n, dd, dd, dd, nd, nd, nd,
                 3, 3, 3, 2, 2, 2, 0, 0, 0);
            conv(S[xi], p.wres, S[o2], p.nin, p.n, dd, dd, dd, nd, nd, nd,
                 3, 3, 3, 2, 2, 2, 0, 0, 0);                            // residual
            float* mnew = mnext[mni++];
            maxmask(mcur, mnew, dd, dd, dd, nd, nd, nd, 3, 3, 3, 2, 2, 2);
            bnrelu(S[o1], mnew, p.bn1g, p.bn1b, p.n, Vout);
            conv(S[o1], p.w2, S[o3], p.n, p.n, nd, nd, nd, nd, nd, nd,
                 3, 3, 3, 1, 1, 1, 1, 1, 1);
            mmul(S[o3], mnew, p.n, Vout);
            addv(S[o3], S[o2], (long)p.n * Vout);
            xi = o3; mcur = mnew; dd = nd;
        }
    }

    // ---- 5. final BN-ReLU + 2x2x2 average ----
    bnrelu(S[xi], mcur, bnfg, bnfb, 256, 8);
    final_pool<<<1, 256, 0, stream>>>(S[xi], (float*)d_out);
}
